// Attention_85976655331618
// MI455X (gfx1250) — compile-verified, matrix operand order checked by reading, not run on hardware
//
#include <hip/hip_runtime.h>
#include <hip/hip_bf16.h>

#define D_MODEL 3072
#define HEADS   24
#define DH      128
#define S_IMG   2048
#define S_TXT   256
#define S_TOT   2304

typedef __attribute__((ext_vector_type(16))) __bf16 bfrag;
typedef __attribute__((ext_vector_type(2)))  __bf16 bf2;
typedef __attribute__((ext_vector_type(8)))  float  f8;
typedef __attribute__((ext_vector_type(2)))  float  f2v;
typedef __attribute__((ext_vector_type(4)))  int    v4i;

union BFU { bfrag v; unsigned short u[16]; };
union U4U { uint4 u4; unsigned short s[8]; };

#if defined(__gfx1250__) && __has_builtin(__builtin_amdgcn_global_load_async_to_lds_b128) && __has_builtin(__builtin_amdgcn_s_wait_asynccnt)
#define USE_ASYNC 1
#endif

// Native scalar f32 -> bf16 (RNE fptrunc; backend picks the cvt instruction).
__device__ __forceinline__ unsigned short f2bf(float x) {
    __bf16 b = (__bf16)x;
    return __builtin_bit_cast(unsigned short, b);
}

// Packed pair conversion: lets the backend select v_cvt_pk_bf16_f32.
__device__ __forceinline__ unsigned int f2bf2(float lo, float hi) {
    f2v f; f[0] = lo; f[1] = hi;
    bf2 b = __builtin_convertvector(f, bf2);
    return __builtin_bit_cast(unsigned int, b);
}

// 16-byte global -> LDS copy; async (no VGPR data path) when available.
// Builtin signature (from compiler diagnostic): (v4i AS1*, v4i AS3*, imm, imm)
__device__ __forceinline__ void cp16_g2l(const unsigned short* g, unsigned short* l) {
#ifdef USE_ASYNC
    __builtin_amdgcn_global_load_async_to_lds_b128(
        (__attribute__((address_space(1))) v4i*)(void*)const_cast<unsigned short*>(g),
        (__attribute__((address_space(3))) v4i*)(void*)l, 0, 0);
#else
    *(uint4*)l = *(const uint4*)g;
#endif
}

__device__ __forceinline__ void cp_wait() {
#ifdef USE_ASYNC
    __builtin_amdgcn_s_wait_asynccnt(0);
#endif
}

// Fragment loader for a tile stored row-major as [rows][K] in LDS (ushort bf16).
// Works for A (rows = M) and for B of an NT GEMM (rows = N), per the CDNA5
// 16-bit matrix VGPR layout: lane&15 selects row, lanes>=16 take K-base 8,
// elements 0..7 = K[kb..kb+7], elements 8..15 = K[kb+16..kb+23].
__device__ __forceinline__ bfrag ld_frag(const unsigned short* base, int row,
                                         int stride, int lane) {
    int m  = lane & 15;
    int kb = (lane & 16) >> 1;   // 0 or 8
    const unsigned short* p = base + (size_t)(row + m) * stride + kb;
    BFU f;
#pragma unroll
    for (int e = 0; e < 8; ++e) { f.u[e] = p[e]; f.u[e + 8] = p[e + 16]; }
    return f.v;
}

__device__ __forceinline__ f8 wmma_bf16(bfrag a, bfrag b, f8 c) {
    return __builtin_amdgcn_wmma_f32_16x16x32_bf16(false, a, false, b,
                                                   (short)0, c, false, false);
}

// ---------------------------------------------------------------------------
// C[M,N] = A[M,K] * W[N,K]^T + bias[N]   (fp32 in/out, bf16 WMMA inside)
// Block: 256 threads (8 waves). Tile 128x128, K-chunk 32, each wave 32x64.
// Software-pipelined: next chunk's global loads issued before the WMMAs.
// ---------------------------------------------------------------------------
#define GK   32
#define LDG  (GK + 8)

__launch_bounds__(256)
__global__ void gemm_nt_bias(const float* __restrict__ A,
                             const float* __restrict__ W,
                             const float* __restrict__ bias,
                             float* __restrict__ C,
                             int M, int N, int K) {
    __shared__ unsigned short sA[128 * LDG];
    __shared__ unsigned short sB[128 * LDG];

    const int tid  = threadIdx.x;
    const int lane = tid & 31;
    const int wid  = tid >> 5;            // 0..7
    const int bm   = blockIdx.y * 128;
    const int bn   = blockIdx.x * 128;
    const int wm   = (wid & 3) * 32;      // 0,32,64,96
    const int wn   = (wid >> 2) * 64;     // 0,64

    f8 acc[2][4];
#pragma unroll
    for (int i = 0; i < 2; ++i)
#pragma unroll
        for (int j = 0; j < 4; ++j) acc[i][j] = {};

    const int r0 = tid >> 3;        // 0..31
    const int cv = (tid & 7) * 4;   // 0..28

    float4 pa[4], pw[4];
#pragma unroll
    for (int i = 0; i < 4; ++i) {
        int row = r0 + 32 * i;
        pa[i] = *(const float4*)(A + (size_t)(bm + row) * K + cv);
        pw[i] = *(const float4*)(W + (size_t)(bn + row) * K + cv);
    }

    for (int kc = 0; kc < K; kc += GK) {
        // packed convert + store the prefetched chunk (2 dword stores each)
#pragma unroll
        for (int i = 0; i < 4; ++i) {
            int row = r0 + 32 * i;
            unsigned int* da = (unsigned int*)(sA + row * LDG + cv);
            da[0] = f2bf2(pa[i].x, pa[i].y);
            da[1] = f2bf2(pa[i].z, pa[i].w);
            unsigned int* dw = (unsigned int*)(sB + row * LDG + cv);
            dw[0] = f2bf2(pw[i].x, pw[i].y);
            dw[1] = f2bf2(pw[i].z, pw[i].w);
        }
        __syncthreads();

        // issue next chunk's global loads (overlap with WMMAs below)
        if (kc + GK < K) {
#pragma unroll
            for (int i = 0; i < 4; ++i) {
                int row = r0 + 32 * i;
                pa[i] = *(const float4*)(A + (size_t)(bm + row) * K + kc + GK + cv);
                pw[i] = *(const float4*)(W + (size_t)(bn + row) * K + kc + GK + cv);
            }
        }

        bfrag a0 = ld_frag(sA, wm,      LDG, lane);
        bfrag a1 = ld_frag(sA, wm + 16, LDG, lane);
        bfrag b[4];
#pragma unroll
        for (int j = 0; j < 4; ++j) b[j] = ld_frag(sB, wn + j * 16, LDG, lane);
#pragma unroll
        for (int j = 0; j < 4; ++j) {
            acc[0][j] = wmma_bf16(a0, b[j], acc[0][j]);
            acc[1][j] = wmma_bf16(a1, b[j], acc[1][j]);
        }
        __syncthreads();
    }

    const int n    = lane & 15;
    const int madd = (lane & 16) >> 1;   // 0 or 8
#pragma unroll
    for (int sm = 0; sm < 2; ++sm)
#pragma unroll
        for (int sn = 0; sn < 4; ++sn) {
            int col = bn + wn + sn * 16 + n;
            float bv = bias[col];
#pragma unroll
            for (int r = 0; r < 8; ++r) {
                int row = bm + wm + sm * 16 + madd + r;
                C[(size_t)row * N + col] = acc[sm][sn][r] + bv;
            }
        }
}

// ---------------------------------------------------------------------------
// Per (token, head): RMSNorm over DH=128 + RoPE, emit bf16 head-major.
// ---------------------------------------------------------------------------
__launch_bounds__(64)
__global__ void norm_rope(const float* __restrict__ src,
                          const float* __restrict__ w,
                          const float* __restrict__ cs,
                          const float* __restrict__ sn,
                          unsigned short* __restrict__ dst,
                          int tokOff) {
    __shared__ float red[64];
    const int p    = threadIdx.x;
    const int tok  = blockIdx.x;
    const int head = blockIdx.y;

    const float* x = src + (size_t)tok * D_MODEL + head * DH;
    float xr = x[2 * p], xi = x[2 * p + 1];
    red[p] = xr * xr + xi * xi;
    __syncthreads();
#pragma unroll
    for (int s = 32; s > 0; s >>= 1) {
        if (p < s) red[p] += red[p + s];
        __syncthreads();
    }
    float inv = rsqrtf(red[0] * (1.0f / 128.0f) + 1e-5f);
    float qr = xr * inv * w[2 * p];
    float qi = xi * inv * w[2 * p + 1];
    float c  = cs[tok * 64 + p];
    float s_ = sn[tok * 64 + p];
    float orr = qr * c - qi * s_;
    float oii = qr * s_ + qi * c;
    unsigned int* o = (unsigned int*)(dst + (size_t)head * S_TOT * DH +
                                      (size_t)(tokOff + tok) * DH + 2 * p);
    *o = f2bf2(orr, oii);
}

// V: fp32 [tok][head][d] -> bf16 head-major [head][tok][d]
__global__ void permute_v(const float* __restrict__ v,
                          unsigned short* __restrict__ dst) {
    int i = blockIdx.x * blockDim.x + threadIdx.x;
    if (i >= S_TOT * D_MODEL) return;
    int d   = i & 127;
    int h   = (i >> 7) % HEADS;
    int tok = i / D_MODEL;
    dst[(size_t)h * S_TOT * DH + (size_t)tok * DH + d] = f2bf(v[i]);
}

__global__ void mask_bias(const unsigned char* __restrict__ mask,
                          float* __restrict__ bias) {
    int i = blockIdx.x * blockDim.x + threadIdx.x;
    if (i >= S_TOT) return;
    bias[i] = (i < S_IMG) ? 0.0f : (mask[i - S_IMG] ? 0.0f : -1e30f);
}

// ---------------------------------------------------------------------------
// Flash attention: grid (36 q-tiles, 24 heads), block 128 (4 waves x 16 rows).
// Q/K tiles arrive via async global->LDS copies (CDNA5 ASYNCcnt path);
// V goes through VGPRs because it is transposed into LDS.
// ---------------------------------------------------------------------------
#define KT 64
#define LK (DH + 8)     // K/Q staging stride (ushort)
#define LV (KT + 8)     // transposed-V stride (ushort)
#define LS (KT + 4)     // S stride (float)
#define LP (KT + 8)     // P stride (ushort)
#define L2E 1.4426950408889634f

__launch_bounds__(128)
__global__ void attn_kernel(const unsigned short* __restrict__ qb,
                            const unsigned short* __restrict__ kb,
                            const unsigned short* __restrict__ vb,
                            const float* __restrict__ kbias,
                            float* __restrict__ attn) {
    __shared__ unsigned short sK[KT * LK];       // K tile (also Q staging)
    __shared__ unsigned short sVt[DH * LV];      // V tile, transposed [d][key]
    __shared__ float          sS[4][16 * LS];
    __shared__ unsigned short sP[4][16 * LP];
    __shared__ float rowM[4][16], rowL[4][16], rowF[4][16];
    __shared__ float pmax[4][2][16], psum[4][2][16];

    const int tid  = threadIdx.x;
    const int lane = tid & 31;
    const int wid  = tid >> 5;
    const int head = blockIdx.y;
    const int qBase = blockIdx.x * 64;

    const unsigned short* qh = qb + (size_t)head * S_TOT * DH;
    const unsigned short* kh = kb + (size_t)head * S_TOT * DH;
    const unsigned short* vh = vb + (size_t)head * S_TOT * DH;

    // ---- stage Q tile through sK (async), build per-wave A-fragments -----
#pragma unroll
    for (int i = 0; i < 8; ++i) {
        int idx = tid + i * 128;
        int row = idx >> 4;
        int c8  = (idx & 15) * 8;
        cp16_g2l(qh + (size_t)(qBase + row) * DH + c8, sK + row * LK + c8);
    }
    cp_wait();
    __syncthreads();
    bfrag aQ[4];
#pragma unroll
    for (int c = 0; c < 4; ++c)
        aQ[c] = ld_frag(sK + 32 * c, wid * 16, LK, lane);
    if (tid < 64) {
        rowM[tid >> 4][tid & 15] = -3e38f;
        rowL[tid >> 4][tid & 15] = 0.0f;
    }
    __syncthreads();

    f8 oAcc[8];
#pragma unroll
    for (int t = 0; t < 8; ++t) oAcc[t] = {};

    const float scale = 0.08838834764831845f;   // 128^-0.5
    const int srow = lane & 15;
    const int half = lane >> 4;
    const int madd = (lane & 16) >> 1;

    for (int kt = 0; kt < S_TOT; kt += KT) {
        // ---- K tile via async copies; V tile transposed through VGPRs ---
#pragma unroll
        for (int i = 0; i < 8; ++i) {
            int idx = tid + i * 128;
            int row = idx >> 4;
            int c8  = (idx & 15) * 8;
            cp16_g2l(kh + (size_t)(kt + row) * DH + c8, sK + row * LK + c8);
            U4U v;
            v.u4 = *(const uint4*)(vh + (size_t)(kt + row) * DH + c8);
#pragma unroll
            for (int e = 0; e < 8; ++e)
                sVt[(c8 + e) * LV + row] = v.s[e];
        }
        cp_wait();
        __syncthreads();

        // ---- S = Q * K^T -------------------------------------------------
        f8 sAcc[4];
#pragma unroll
        for (int j = 0; j < 4; ++j) sAcc[j] = {};
#pragma unroll
        for (int j = 0; j < 4; ++j)
#pragma unroll
            for (int c = 0; c < 4; ++c) {
                bfrag bk = ld_frag(sK + 32 * c, j * 16, LK, lane);
                sAcc[j] = wmma_bf16(aQ[c], bk, sAcc[j]);
            }
        // scale + mask bias, spill to LDS for cross-lane softmax
        {
            int n = lane & 15;
#pragma unroll
            for (int j = 0; j < 4; ++j) {
                float bv = kbias[kt + j * 16 + n];
#pragma unroll
                for (int r = 0; r < 8; ++r)
                    sS[wid][(madd + r) * LS + j * 16 + n] =
                        sAcc[j][r] * scale + bv;
            }
        }
        __syncthreads();

        // ---- online softmax ---------------------------------------------
        float mx = -3e38f;
#pragma unroll
        for (int c2 = 0; c2 < 32; ++c2)
            mx = fmaxf(mx, sS[wid][srow * LS + half * 32 + c2]);
        pmax[wid][half][srow] = mx;
        __syncthreads();
        if (lane < 16) {
            float mOld = rowM[wid][srow];
            float mNew = fmaxf(mOld, fmaxf(pmax[wid][0][srow], pmax[wid][1][srow]));
            rowF[wid][srow] = exp2f((mOld - mNew) * L2E);
            rowM[wid][srow] = mNew;
        }
        __syncthreads();

        float fr[8];
#pragma unroll
        for (int r = 0; r < 8; ++r) fr[r] = rowF[wid][madd + r];
#pragma unroll
        for (int t = 0; t < 8; ++t)
#pragma unroll
            for (int r = 0; r < 8; ++r) oAcc[t][r] *= fr[r];

        float mNew = rowM[wid][srow];
        float lsum = 0.0f;
        {
            unsigned int* prow =
                (unsigned int*)(sP[wid] + srow * LP + half * 32);
#pragma unroll
            for (int c2 = 0; c2 < 32; c2 += 2) {
                float s0 = sS[wid][srow * LS + half * 32 + c2];
                float s1 = sS[wid][srow * LS + half * 32 + c2 + 1];
                float p0 = exp2f((s0 - mNew) * L2E);
                float p1 = exp2f((s1 - mNew) * L2E);
                prow[c2 >> 1] = f2bf2(p0, p1);
                lsum += p0 + p1;
            }
        }
        psum[wid][half][srow] = lsum;
        __syncthreads();
        if (lane < 16)
            rowL[wid][srow] = rowL[wid][srow] * rowF[wid][srow] +
                              psum[wid][0][srow] + psum[wid][1][srow];

        // ---- O += P * V --------------------------------------------------
        bfrag aP[2];
        aP[0] = ld_frag(sP[wid],      0, LP, lane);
        aP[1] = ld_frag(sP[wid] + 32, 0, LP, lane);
#pragma unroll
        for (int t = 0; t < 8; ++t)
#pragma unroll
            for (int c = 0; c < 2; ++c) {
                bfrag bv = ld_frag(sVt + 32 * c, t * 16, LV, lane);
                oAcc[t] = wmma_bf16(aP[c], bv, oAcc[t]);
            }
        __syncthreads();
    }

    // ---- epilogue: divide by row sums, write [tok][head*128+d] -----------
    float lr[8];
#pragma unroll
    for (int r = 0; r < 8; ++r) lr[r] = rowL[wid][madd + r];
    int n = lane & 15;
#pragma unroll
    for (int t = 0; t < 8; ++t)
#pragma unroll
        for (int r = 0; r < 8; ++r) {
            int row = qBase + wid * 16 + madd + r;
            attn[(size_t)row * D_MODEL + head * DH + t * 16 + n] =
                oAcc[t][r] / lr[r];
        }
}

// ---------------------------------------------------------------------------
extern "C" void kernel_launch(void* const* d_in, const int* in_sizes, int n_in,
                              void* d_out, int out_size, void* d_ws, size_t ws_size,
                              hipStream_t stream) {
    const float* hid   = (const float*)d_in[0];
    const float* enc   = (const float*)d_in[1];
    const unsigned char* msk = (const unsigned char*)d_in[2];
    const float* icos  = (const float*)d_in[3];
    const float* isin  = (const float*)d_in[4];
    const float* tcos  = (const float*)d_in[5];
    const float* tsin  = (const float*)d_in[6];
    const float* wq    = (const float*)d_in[7];
    const float* bq    = (const float*)d_in[8];
    const float* wk    = (const float*)d_in[9];
    const float* bk    = (const float*)d_in[10];
    const float* wv    = (const float*)d_in[11];
    const float* bv    = (const float*)d_in[12];
    const float* awq   = (const float*)d_in[13];
    const float* abq   = (const float*)d_in[14];
    const float* awk   = (const float*)d_in[15];
    const float* abk   = (const float*)d_in[16];
    const float* awv   = (const float*)d_in[17];
    const float* abv   = (const float*)d_in[18];
    const float* nqw   = (const float*)d_in[19];
    const float* nkw   = (const float*)d_in[20];
    const float* naqw  = (const float*)d_in[21];
    const float* nakw  = (const float*)d_in[22];
    const float* wout  = (const float*)d_in[23];
    const float* bout  = (const float*)d_in[24];
    const float* waout = (const float*)d_in[25];
    const float* baout = (const float*)d_in[26];
    float* out = (float*)d_out;

    const size_t SD = (size_t)S_TOT * D_MODEL;      // 7,077,888
    char* ws = (char*)d_ws;
    float* q_f32 = (float*)ws;                 ws += SD * sizeof(float);
    float* k_f32 = (float*)ws;                 ws += SD * sizeof(float);
    float* v_f32 = (float*)ws;                 ws += SD * sizeof(float);
    float* attnb = (float*)ws;                 ws += SD * sizeof(float);
    float* biasb = (float*)ws;                 ws += 4096;
    unsigned short* qb   = (unsigned short*)ws; ws += SD * sizeof(unsigned short);
    unsigned short* kbuf = (unsigned short*)ws; ws += SD * sizeof(unsigned short);
    unsigned short* vbuf = (unsigned short*)ws; ws += SD * sizeof(unsigned short);

    dim3 blk(256);
    dim3 gI(D_MODEL / 128, S_IMG / 128);   // 24 x 16
    dim3 gT(D_MODEL / 128, S_TXT / 128);   // 24 x 2

    mask_bias<<<(S_TOT + 255) / 256, 256, 0, stream>>>(msk, biasb);

    // QKV projections
    gemm_nt_bias<<<gI, blk, 0, stream>>>(hid, wq, bq, q_f32, S_IMG, D_MODEL, D_MODEL);
    gemm_nt_bias<<<gI, blk, 0, stream>>>(hid, wk, bk, k_f32, S_IMG, D_MODEL, D_MODEL);
    gemm_nt_bias<<<gI, blk, 0, stream>>>(hid, wv, bv, v_f32, S_IMG, D_MODEL, D_MODEL);
    gemm_nt_bias<<<gT, blk, 0, stream>>>(enc, awq, abq, q_f32 + (size_t)S_IMG * D_MODEL, S_TXT, D_MODEL, D_MODEL);
    gemm_nt_bias<<<gT, blk, 0, stream>>>(enc, awk, abk, k_f32 + (size_t)S_IMG * D_MODEL, S_TXT, D_MODEL, D_MODEL);
    gemm_nt_bias<<<gT, blk, 0, stream>>>(enc, awv, abv, v_f32 + (size_t)S_IMG * D_MODEL, S_TXT, D_MODEL, D_MODEL);

    // RMSNorm + RoPE -> bf16 head-major
    norm_rope<<<dim3(S_IMG, HEADS), 64, 0, stream>>>(q_f32, nqw, icos, isin, qb, 0);
    norm_rope<<<dim3(S_IMG, HEADS), 64, 0, stream>>>(k_f32, nkw, icos, isin, kbuf, 0);
    norm_rope<<<dim3(S_TXT, HEADS), 64, 0, stream>>>(q_f32 + (size_t)S_IMG * D_MODEL, naqw, tcos, tsin, qb, S_IMG);
    norm_rope<<<dim3(S_TXT, HEADS), 64, 0, stream>>>(k_f32 + (size_t)S_IMG * D_MODEL, nakw, tcos, tsin, kbuf, S_IMG);
    permute_v<<<(int)((SD + 255) / 256), 256, 0, stream>>>(v_f32, vbuf);

    // Flash attention
    attn_kernel<<<dim3(S_TOT / 64, HEADS), 128, 0, stream>>>(qb, kbuf, vbuf, biasb, attnb);

    // Output projections (tuple layout: img then txt)
    gemm_nt_bias<<<gI, blk, 0, stream>>>(attnb, wout, bout, out, S_IMG, D_MODEL, D_MODEL);
    gemm_nt_bias<<<gT, blk, 0, stream>>>(attnb + (size_t)S_IMG * D_MODEL, waout, baout,
                                         out + (size_t)S_IMG * D_MODEL, S_TXT, D_MODEL, D_MODEL);
}
